// WanSelfAttention_7258494730375
// MI455X (gfx1250) — compile-verified, matrix operand order checked by reading, not run on hardware
//
#include <hip/hip_runtime.h>
#include <math.h>

// ---------------- problem constants (from reference) ----------------
constexpr int kDIM = 1536;
constexpr int kNH  = 12;
constexpr int kHD  = 128;          // head dim
constexpr int kC   = 64;           // rope pairs per head
constexpr int kF = 6, kH = 18, kW = 32;
constexpr int kS   = kF * kH * kW; // 3456
constexpr float kEPS   = 1e-6f;
constexpr float kSCALE = 0.08838834764831845f; // 1/sqrt(128)
// rope splits over the 64 pairs: (22, 21, 21) -> boundaries 22, 43
constexpr int kSPLIT0 = 22, kSPLIT1 = 43;

typedef __attribute__((ext_vector_type(16))) _Float16 v16h;
typedef __attribute__((ext_vector_type(8)))  _Float16 v8h;
typedef __attribute__((ext_vector_type(8)))  float    v8f;
typedef __attribute__((ext_vector_type(4)))  unsigned int u32x4;
typedef __attribute__((ext_vector_type(8)))  unsigned int u32x8;

// ---------------- WMMA helpers ----------------
// 16-bit A/B fragment for v_wmma_f32_16x16x32_f16.
// Per ISA: lane<16 holds K {0..7} and {16..23}; lane>=16 holds K {8..15} and {24..31}.
__device__ inline v16h load_frag_row(const _Float16* rowp, int half, int k0) {
  v8h lo = *(const v8h*)(rowp + k0 + 8 * half);
  v8h hi = *(const v8h*)(rowp + k0 + 16 + 8 * half);
  return __builtin_shufflevector(lo, hi, 0,1,2,3,4,5,6,7,8,9,10,11,12,13,14,15);
}

__device__ inline v8f wmma_f16(v16h a, v16h b, v8f c) {
  return __builtin_amdgcn_wmma_f32_16x16x32_f16(
      /*neg_a=*/false, a, /*neg_b=*/false, b,
      /*c_mod=*/(short)0, c, /*reuse_a=*/false, /*reuse_b=*/false);
}

// ---------------- async global->LDS (ASYNCcnt path) ----------------
// copies 16 bytes per lane from global memory to LDS, tracked by ASYNCcnt.
__device__ inline void async_cp16(unsigned lds_addr, const void* gaddr) {
  asm volatile("global_load_async_to_lds_b128 %0, %1, off"
               :: "v"(lds_addr), "v"(gaddr) : "memory");
}
__device__ inline void async_cp16_off16(unsigned lds_addr, const void* gaddr) {
  asm volatile("global_load_async_to_lds_b128 %0, %1, off offset:16"
               :: "v"(lds_addr), "v"(gaddr) : "memory");
}
__device__ inline void wait_asynccnt0() {
  asm volatile("s_wait_asynccnt 0x0" ::: "memory");
}

// ---------------- Tensor Data Mover: 2D tile load (TENSORcnt path) ----------------
// D# per CDNA5 ISA; 2-group form (<=2D tensor).
// All sizes/strides in elements of data_size = 2 bytes.
__device__ inline void tdm_load_2d(unsigned lds_addr, const void* gptr,
                                   unsigned tensor_d0, unsigned tensor_d1,
                                   unsigned tile_d0, unsigned tile_d1,
                                   unsigned long long stride0) {
  const unsigned long long ga = (unsigned long long)(uintptr_t)gptr;
  u32x4 g0;
  g0.x = 1u;                                         // count=1, is_restore=0
  g0.y = lds_addr;                                   // LDS byte address
  g0.z = (unsigned)(ga & 0xffffffffu);               // global_addr[31:0]
  g0.w = (unsigned)((ga >> 32) & 0x1ffffffu)         // global_addr[56:32]
         | (2u << 30);                               // type = 2 ("image")
  u32x8 g1;
  g1.s0 = 1u << 16;                                  // data_size=1 -> 2 bytes
  g1.s1 = (tensor_d0 & 0xffffu) << 16;               // tensor_dim0[15:0]
  g1.s2 = (tensor_d0 >> 16) | ((tensor_d1 & 0xffffu) << 16);
  g1.s3 = (tensor_d1 >> 16) | ((tile_d0 & 0xffffu) << 16);
  g1.s4 = tile_d1 & 0xffffu;                         // tile_dim1 (tile_dim2=0)
  g1.s5 = (unsigned)(stride0 & 0xffffffffu);         // tensor_dim0_stride[31:0]
  g1.s6 = (unsigned)((stride0 >> 32) & 0xffffu);     // stride[47:32], dim1_stride lo=0
  g1.s7 = 0u;
  asm volatile("tensor_load_to_lds %0, %1" :: "s"(g0), "s"(g1) : "memory");
}

// ---------------- f32 -> f16 convert ----------------
__global__ __launch_bounds__(256) void cvt_f32_f16(const float* __restrict__ s,
                                                   _Float16* __restrict__ d, int n) {
  int i = blockIdx.x * 256 + threadIdx.x;
  if (i < n) d[i] = (_Float16)s[i];
}

// ---------------- WMMA GEMM: C[M,N] = A[M,K](f16) * Bw[N,K](f16)^T + bias ----------------
// Bw rows (length K) are exactly the per-lane B^T fragments.
// Block tile 128x128, 8 waves (4 along M, 2 along N), per-wave 32x64.
// Staging uses global_load_async_to_lds_b128 (ASYNCcnt).
__global__ __launch_bounds__(256) void gemm_f16_wmma(
    const _Float16* __restrict__ A, const _Float16* __restrict__ Bw,
    const float* __restrict__ bias, float* __restrict__ Cout,
    int M, int N, int K) {
  __shared__ _Float16 As[128 * 32];
  __shared__ _Float16 Bs[128 * 32];
  const int t = threadIdx.x;
  const int w = t >> 5, lane = t & 31, half = lane >> 4, l16 = lane & 15;
  const int wm = w & 3, wn = w >> 2;                  // wave position in 4x2 grid
  const int mblk = blockIdx.y * 128, nblk = blockIdx.x * 128;

  v8f acc[2][4];
#pragma unroll
  for (int i = 0; i < 2; i++)
#pragma unroll
    for (int j = 0; j < 4; j++) acc[i][j] = (v8f){};

  const int nchunk = K / 32;
  const int lr = t >> 1, lsg = (t & 1) * 16;          // 128 rows x two 32B halves
  const unsigned la = (unsigned)(uintptr_t)(As + lr * 32 + lsg);
  const unsigned lb = (unsigned)(uintptr_t)(Bs + lr * 32 + lsg);
  for (int kc = 0; kc < nchunk; kc++) {
    { // async global->LDS staging (32-wide k chunk), 32B of A + 32B of B per lane
      const _Float16* ga = A  + (size_t)(mblk + lr) * K + kc * 32 + lsg;
      const _Float16* gb = Bw + (size_t)(nblk + lr) * K + kc * 32 + lsg;
      async_cp16(la, ga);
      async_cp16_off16(la, ga);
      async_cp16(lb, gb);
      async_cp16_off16(lb, gb);
      if (kc + 1 < nchunk) {                          // prefetch next chunk
        __builtin_prefetch(ga + 32, 0, 1);
        __builtin_prefetch(gb + 32, 0, 1);
      }
    }
    wait_asynccnt0();
    __syncthreads();
    v16h af[2], bf[4];
#pragma unroll
    for (int i = 0; i < 2; i++)
      af[i] = load_frag_row(As + (wm * 32 + i * 16 + l16) * 32, half, 0);
#pragma unroll
    for (int j = 0; j < 4; j++)
      bf[j] = load_frag_row(Bs + (wn * 64 + j * 16 + l16) * 32, half, 0);
#pragma unroll
    for (int i = 0; i < 2; i++)
#pragma unroll
      for (int j = 0; j < 4; j++) acc[i][j] = wmma_f16(af[i], bf[j], acc[i][j]);
    __syncthreads();
  }

  // epilogue: accumulator element r is (M = r + 8*half, N = l16) within a 16x16 tile
#pragma unroll
  for (int i = 0; i < 2; i++)
#pragma unroll
    for (int j = 0; j < 4; j++) {
      const int col = nblk + wn * 64 + j * 16 + l16;
      const float bv = bias[col];
#pragma unroll
      for (int r = 0; r < 8; r++) {
        const int row = mblk + wm * 32 + i * 16 + r + 8 * half;
        Cout[(size_t)row * N + col] = acc[i][j][r] + bv;
      }
    }
}

// ---------------- fused RMSNorm + 3D RoPE, f32 [S][DIM] -> f16 [NH][S][HD] ----------------
__global__ __launch_bounds__(256) void rmsnorm_rope(
    const float* __restrict__ X, const float* __restrict__ g,
    const float* __restrict__ fcos, const float* __restrict__ fsin,
    _Float16* __restrict__ Out) {
  const int s = blockIdx.x, t = threadIdx.x;
  const float* row = X + (size_t)s * kDIM;
  __shared__ float red[256];
  float ss = 0.f;
  for (int c = t; c < kDIM; c += 256) { float v = row[c]; ss += v * v; }
  red[t] = ss;
  __syncthreads();
  for (int o = 128; o > 0; o >>= 1) {
    if (t < o) red[t] += red[t + o];
    __syncthreads();
  }
  const float inv = rsqrtf(red[0] / (float)kDIM + kEPS);

  const int f  = s / (kH * kW);
  const int hc = (s / kW) % kH;
  const int wc = s % kW;

  for (int p = t; p < kDIM / 2; p += 256) {   // p = global pair index
    const int head = p / (kHD / 2);
    const int pc   = p % (kHD / 2);           // pair within head, 0..63
    const int coord = (pc < kSPLIT0) ? f : ((pc < kSPLIT1) ? hc : wc);
    const float cs = fcos[coord * kC + pc];
    const float sn = fsin[coord * kC + pc];
    const float re = row[2 * p]     * inv * g[2 * p];
    const float im = row[2 * p + 1] * inv * g[2 * p + 1];
    const size_t base = ((size_t)head * kS + s) * kHD + 2 * pc;
    Out[base]     = (_Float16)(re * cs - im * sn);
    Out[base + 1] = (_Float16)(re * sn + im * cs);
  }
}

// ---------------- V f32 [S][DIM] -> Vt f16 [NH][HD][S] ----------------
__global__ __launch_bounds__(256) void v_to_vt(const float* __restrict__ V,
                                               _Float16* __restrict__ Vt) {
  int i = blockIdx.x * 256 + threadIdx.x;
  if (i >= kS * kDIM) return;
  const int s = i / kDIM, c = i % kDIM;
  const int h = c / kHD, d = c % kHD;
  Vt[((size_t)h * kHD + d) * kS + s] = (_Float16)V[i];
}

// ---------------- flash attention ----------------
// grid (S/128, NH); 8 waves; each wave owns 16 query rows; 32-key chunks.
// K/V chunks staged by the Tensor Data Mover (TENSORcnt), issued by wave 0.
__global__ __launch_bounds__(256) void attn_wmma(
    const _Float16* __restrict__ Qh,  // [NH][S][HD]
    const _Float16* __restrict__ Kh,  // [NH][S][HD]
    const _Float16* __restrict__ Vt,  // [NH][HD][S]
    _Float16* __restrict__ Oh) {      // [S][DIM]
  __shared__ _Float16 Ks[32 * kHD];   // [32 keys][128]
  __shared__ _Float16 Vs[kHD * 32];   // [128 d][32 keys]
  __shared__ _Float16 Ps[8 * 16 * 32];// per-wave P tile [16 q][32 keys]
  const int t = threadIdx.x, w = t >> 5, lane = t & 31;
  const int half = lane >> 4, l16 = lane & 15;
  const int h = blockIdx.y;
  const int qbase = blockIdx.x * 128 + w * 16;
  const _Float16* Qhh = Qh + (size_t)h * kS * kHD;
  const _Float16* Khh = Kh + (size_t)h * kS * kHD;
  const _Float16* Vth = Vt + (size_t)h * kHD * kS;
  _Float16* pw = Ps + w * 512;

  // preload Q fragments: 16 rows x 128 k  ->  4 A-frags
  v16h aq[4];
  {
    const _Float16* qrow = Qhh + (size_t)(qbase + l16) * kHD;
#pragma unroll
    for (int kk = 0; kk < 4; kk++) aq[kk] = load_frag_row(qrow, half, kk * 32);
  }

  v8f oacc[8];
  float mrow[8], lrow[8];
#pragma unroll
  for (int j = 0; j < 8; j++) oacc[j] = (v8f){};
#pragma unroll
  for (int i = 0; i < 8; i++) { mrow[i] = -3.0e38f; lrow[i] = 0.f; }

  const unsigned ldsK = (unsigned)(uintptr_t)Ks;
  const unsigned ldsV = (unsigned)(uintptr_t)Vs;

  for (int kc = 0; kc < kS / 32; kc++) {
    if (w == 0) { // TDM: stage K (32x128, contiguous) and Vt (128x32, strided)
      tdm_load_2d(ldsK, Khh + (size_t)kc * 32 * kHD,
                  /*tensor_d0=*/kHD, /*tensor_d1=*/kS,
                  /*tile_d0=*/kHD, /*tile_d1=*/32, /*stride0=*/kHD);
      tdm_load_2d(ldsV, Vth + (size_t)kc * 32,
                  /*tensor_d0=*/kS, /*tensor_d1=*/kHD,
                  /*tile_d0=*/32, /*tile_d1=*/kHD, /*stride0=*/kS);
      __builtin_amdgcn_s_wait_tensorcnt(0);
    }
    __syncthreads();

    // scores: two 16x16 tiles (keys 0-15, 16-31), k = HD = 4 x 32
    v8f sc[2];
#pragma unroll
    for (int tt = 0; tt < 2; tt++) {
      v8f a = (v8f){};
#pragma unroll
      for (int kk = 0; kk < 4; kk++) {
        v16h bf = load_frag_row(Ks + (tt * 16 + l16) * kHD, half, kk * 32);
        a = wmma_f16(aq[kk], bf, a);
      }
      sc[tt] = a;
    }
#pragma unroll
    for (int tt = 0; tt < 2; tt++)
#pragma unroll
      for (int i = 0; i < 8; i++) sc[tt][i] *= kSCALE;

    // online softmax: row i lives in accumulator slot i across the 16 lanes of a half
    float alpha[8];
#pragma unroll
    for (int i = 0; i < 8; i++) {
      float v = fmaxf(sc[0][i], sc[1][i]);
#pragma unroll
      for (int msk = 1; msk < 16; msk <<= 1) v = fmaxf(v, __shfl_xor(v, msk, 32));
      const float mn = fmaxf(mrow[i], v);
      alpha[i] = __expf(mrow[i] - mn);
      mrow[i] = mn;
    }
#pragma unroll
    for (int i = 0; i < 8; i++) {
      const float p0 = __expf(sc[0][i] - mrow[i]);
      const float p1 = __expf(sc[1][i] - mrow[i]);
      sc[0][i] = p0; sc[1][i] = p1;
      float v = p0 + p1;
#pragma unroll
      for (int msk = 1; msk < 16; msk <<= 1) v += __shfl_xor(v, msk, 32);
      lrow[i] = lrow[i] * alpha[i] + v;
    }

    // re-layout P (accumulator layout -> A-fragment layout) through per-wave LDS
#pragma unroll
    for (int tt = 0; tt < 2; tt++)
#pragma unroll
      for (int i = 0; i < 8; i++)
        pw[(i + 8 * half) * 32 + tt * 16 + l16] = (_Float16)sc[tt][i];
    asm volatile("s_wait_dscnt 0" ::: "memory");   // wave-local LDS RAW fence
    v16h pf = load_frag_row(pw + l16 * 32, half, 0);

    // rescale O and accumulate P x V  (HD = 8 tiles of 16)
#pragma unroll
    for (int j = 0; j < 8; j++)
#pragma unroll
      for (int i = 0; i < 8; i++) oacc[j][i] *= alpha[i];
#pragma unroll
    for (int j = 0; j < 8; j++) {
      v16h vf = load_frag_row(Vs + (j * 16 + l16) * 32, half, 0);
      oacc[j] = wmma_f16(pf, vf, oacc[j]);
    }
    __syncthreads();
  }

  // epilogue: normalize and scatter to merged-head layout [S][DIM]
#pragma unroll
  for (int i = 0; i < 8; i++) lrow[i] = 1.0f / lrow[i];
#pragma unroll
  for (int j = 0; j < 8; j++)
#pragma unroll
    for (int i = 0; i < 8; i++) {
      const int row = qbase + i + 8 * half;
      const int col = h * kHD + j * 16 + l16;
      Oh[(size_t)row * kDIM + col] = (_Float16)(oacc[j][i] * lrow[i]);
    }
}

// ---------------- host launch ----------------
extern "C" void kernel_launch(void* const* d_in, const int* in_sizes, int n_in,
                              void* d_out, int out_size, void* d_ws, size_t ws_size,
                              hipStream_t stream) {
  (void)in_sizes; (void)n_in; (void)out_size; (void)ws_size;
  const float* x    = (const float*)d_in[0];
  const float* fcos = (const float*)d_in[1];
  const float* fsin = (const float*)d_in[2];
  const float* Wq = (const float*)d_in[3];  const float* bq = (const float*)d_in[4];
  const float* Wk = (const float*)d_in[5];  const float* bk = (const float*)d_in[6];
  const float* Wv = (const float*)d_in[7];  const float* bv = (const float*)d_in[8];
  const float* Wo = (const float*)d_in[9];  const float* bo = (const float*)d_in[10];
  const float* gq = (const float*)d_in[11]; const float* gk = (const float*)d_in[12];

  char* p = (char*)d_ws;
  size_t off = 0;
  auto alloc = [&](size_t bytes) -> void* {
    void* r = p + off;
    off = (off + bytes + 255) & ~(size_t)255;
    return r;
  };
  _Float16* x16  = (_Float16*)alloc((size_t)kS * kDIM * 2);
  _Float16* Wq16 = (_Float16*)alloc((size_t)kDIM * kDIM * 2);
  _Float16* Wk16 = (_Float16*)alloc((size_t)kDIM * kDIM * 2);
  _Float16* Wv16 = (_Float16*)alloc((size_t)kDIM * kDIM * 2);
  _Float16* Wo16 = (_Float16*)alloc((size_t)kDIM * kDIM * 2);
  float*    Qf   = (float*)   alloc((size_t)kS * kDIM * 4);
  float*    Kf   = (float*)   alloc((size_t)kS * kDIM * 4);
  float*    Vf   = (float*)   alloc((size_t)kS * kDIM * 4);
  _Float16* Qh   = (_Float16*)alloc((size_t)kNH * kS * kHD * 2);
  _Float16* Kh   = (_Float16*)alloc((size_t)kNH * kS * kHD * 2);
  _Float16* Vth  = (_Float16*)alloc((size_t)kNH * kHD * kS * 2);
  _Float16* Oh   = (_Float16*)alloc((size_t)kS * kDIM * 2);

  const int nx = kS * kDIM, nw = kDIM * kDIM;
  cvt_f32_f16<<<(nx + 255) / 256, 256, 0, stream>>>(x,  x16,  nx);
  cvt_f32_f16<<<(nw + 255) / 256, 256, 0, stream>>>(Wq, Wq16, nw);
  cvt_f32_f16<<<(nw + 255) / 256, 256, 0, stream>>>(Wk, Wk16, nw);
  cvt_f32_f16<<<(nw + 255) / 256, 256, 0, stream>>>(Wv, Wv16, nw);
  cvt_f32_f16<<<(nw + 255) / 256, 256, 0, stream>>>(Wo, Wo16, nw);

  dim3 gsz(kDIM / 128, kS / 128);   // (12, 27)
  gemm_f16_wmma<<<gsz, 256, 0, stream>>>(x16, Wq16, bq, Qf, kS, kDIM, kDIM);
  gemm_f16_wmma<<<gsz, 256, 0, stream>>>(x16, Wk16, bk, Kf, kS, kDIM, kDIM);
  gemm_f16_wmma<<<gsz, 256, 0, stream>>>(x16, Wv16, bv, Vf, kS, kDIM, kDIM);

  rmsnorm_rope<<<kS, 256, 0, stream>>>(Qf, gq, fcos, fsin, Qh);
  rmsnorm_rope<<<kS, 256, 0, stream>>>(Kf, gk, fcos, fsin, Kh);
  v_to_vt<<<(nx + 255) / 256, 256, 0, stream>>>(Vf, Vth);

  dim3 ga(kS / 128, kNH);           // (27, 12)
  attn_wmma<<<ga, 256, 0, stream>>>(Qh, Kh, Vth, Oh);

  gemm_f16_wmma<<<gsz, 256, 0, stream>>>(Oh, Wo16, bo, (float*)d_out, kS, kDIM, kDIM);
}
